// LocalMovePolicy_3178275799164
// MI455X (gfx1250) — matrix operand order repaired and checked by reading, compiled, not run on hardware
//
#include <hip/hip_runtime.h>
#include <hip/hip_bf16.h>

#define NN    20000
#define EE    320000
#define FIN   5
#define HH    128
#define HEADS 4
#define LL    3
#define SS    10000
#define KDIM  512          // HEADS*HH
#define NEG_SLOPE 0.2f
#define LN_EPS 1e-5f

typedef __attribute__((ext_vector_type(16))) __bf16 v16bf;
typedef __attribute__((ext_vector_type(8)))  float  v8f;

union FragBF { v16bf v; unsigned int u32[8]; };

__device__ __forceinline__ unsigned short f2bf(float f) {
    unsigned int u = __float_as_uint(f);
    unsigned int r = (u + 0x7fffu + ((u >> 16) & 1u)) >> 16;  // RNE
    return (unsigned short)r;
}
__device__ __forceinline__ unsigned int fenc(float f) {       // order-preserving f32->u32
    unsigned int u = __float_as_uint(f);
    return (u & 0x80000000u) ? ~u : (u | 0x80000000u);
}
__device__ __forceinline__ float fdec(unsigned int u) {
    unsigned int v = (u & 0x80000000u) ? (u & 0x7FFFFFFFu) : ~u;
    return __uint_as_float(v);
}

// ---- K0: lin_W [L,128,512] f32 -> Wt [L,512,128] bf16 (B col-major) ----
__global__ void k_convert_w(const float* __restrict__ linW, unsigned short* __restrict__ wt) {
    int t = blockIdx.x * blockDim.x + threadIdx.x;
    if (t >= LL * KDIM * HH) return;
    int l = t / (KDIM * HH);
    int r = t - l * (KDIM * HH);
    int n = r >> 7;          // 0..511 output column
    int k = r & 127;         // 0..127 reduction index
    wt[t] = f2bf(linW[l * HH * KDIM + k * KDIM + n]);
}

// ---- K1: node embedding: h = relu(x @ Wn + bn) ----
__global__ void k_node_embed(const float* __restrict__ x, const float* __restrict__ Wn,
                             const float* __restrict__ bn, float* __restrict__ h,
                             unsigned short* __restrict__ hbf) {
    int t = blockIdx.x * blockDim.x + threadIdx.x;
    if (t >= NN * HH) return;
    int n = t >> 7, c = t & 127;
    float acc = bn[c];
#pragma unroll
    for (int f = 0; f < FIN; ++f) acc = fmaf(x[n * FIN + f], Wn[f * HH + c], acc);
    acc = fmaxf(acc, 0.f);
    h[t] = acc;
    hbf[t] = f2bf(acc);
}

// ---- K2: WMMA GEMM  xp[N,512] = h_bf16[N,128] @ Wt_bf16[512,128]^T ----
// One wave per 16x64 output tile: A fragment reused across 4 accumulators,
// 16x v_wmma_f32_16x16x32_bf16 per wave in 4 independent back-to-back groups.
__global__ void k_gemm_wmma(const unsigned short* __restrict__ hbf,
                            const unsigned short* __restrict__ wt,   // [512][128]
                            float* __restrict__ xp) {
    int wave = (blockIdx.x * blockDim.x + threadIdx.x) >> 5;
    int lane = threadIdx.x & 31;
    int colGrp  = wave & 7;         // 512/64 = 8 column groups
    int rowTile = wave >> 3;        // 20000/16 = 1250 row tiles
    int half = lane >> 4;
    int l16  = lane & 15;

    const unsigned int* arow  = (const unsigned int*)(hbf + (size_t)(rowTile * 16 + l16) * HH);
    const unsigned int* bbase = (const unsigned int*)(wt  + (size_t)(colGrp * 64 + l16) * HH);
    const int colStride = 16 * (HH / 2);   // dwords between 16-column subtiles

    v8f acc0 = {}, acc1 = {}, acc2 = {}, acc3 = {};
#pragma unroll
    for (int kc = 0; kc < 4; ++kc) {
        int k0 = kc * 32;
        FragBF a, b0, b1, b2, b3;
#pragma unroll
        for (int p = 0; p < 8; ++p) {
            // A 16-bit 16x32 layout: pair base K = half*16 + p*2 (contiguous, matches B)
            int kb = k0 + half * 16 + p * 2;
            a.u32[p] = arow[kb >> 1];
        }
        int boff = (k0 + half * 16) >> 1;  // 16 contiguous bf16 per lane
#pragma unroll
        for (int p = 0; p < 8; ++p) b0.u32[p] = bbase[boff + p];
#pragma unroll
        for (int p = 0; p < 8; ++p) b1.u32[p] = bbase[colStride + boff + p];
#pragma unroll
        for (int p = 0; p < 8; ++p) b2.u32[p] = bbase[2 * colStride + boff + p];
#pragma unroll
        for (int p = 0; p < 8; ++p) b3.u32[p] = bbase[3 * colStride + boff + p];
        acc0 = __builtin_amdgcn_wmma_f32_16x16x32_bf16(false, a.v, false, b0.v,
                                                       (short)0, acc0, false, false);
        acc1 = __builtin_amdgcn_wmma_f32_16x16x32_bf16(false, a.v, false, b1.v,
                                                       (short)0, acc1, false, false);
        acc2 = __builtin_amdgcn_wmma_f32_16x16x32_bf16(false, a.v, false, b2.v,
                                                       (short)0, acc2, false, false);
        acc3 = __builtin_amdgcn_wmma_f32_16x16x32_bf16(false, a.v, false, b3.v,
                                                       (short)0, acc3, false, false);
    }
    float* orow = xp + (size_t)(rowTile * 16) * KDIM + colGrp * 64 + l16;
#pragma unroll
    for (int r = 0; r < 8; ++r) {
        int m = r + half * 8;                       // C/D layout: lanes16-31 -> M+8
        orow[(size_t)m * KDIM + 0]  = acc0[r];
        orow[(size_t)m * KDIM + 16] = acc1[r];
        orow[(size_t)m * KDIM + 32] = acc2[r];
        orow[(size_t)m * KDIM + 48] = acc3[r];
    }
}

// ---- K3: per-(node,head) attention dots ----
__global__ void k_alphas(const float* __restrict__ xp, const float* __restrict__ as,
                         const float* __restrict__ ad, float* __restrict__ aS,
                         float* __restrict__ aD) {
    int t = blockIdx.x * blockDim.x + threadIdx.x;
    if (t >= NN * HEADS) return;
    int n = t >> 2, hd = t & 3;
    const float* xr = xp + (size_t)n * KDIM + hd * HH;
    const float* s  = as + hd * HH;
    const float* d  = ad + hd * HH;
    float vs = 0.f, vd = 0.f;
#pragma unroll 4
    for (int c = 0; c < HH; ++c) { vs = fmaf(xr[c], s[c], vs); vd = fmaf(xr[c], d[c], vd); }
    aS[t] = vs; aD[t] = vd;
}

// ---- K3b: init segment-softmax state + agg ----
__global__ void k_init(unsigned int* __restrict__ emax, float* __restrict__ esum,
                       float* __restrict__ agg) {
    int t = blockIdx.x * blockDim.x + threadIdx.x;
    if (t < NN * HEADS) { emax[t] = 0x007FFFFFu; /* enc(-inf) */ esum[t] = 0.f; }
    if (t < NN * HH) agg[t] = 0.f;
}

// ---- K4: segment max over incoming edges ----
__global__ void k_edge_max(const int* __restrict__ ei, const float* __restrict__ aS,
                           const float* __restrict__ aD, unsigned int* __restrict__ emax) {
    int t = blockIdx.x * blockDim.x + threadIdx.x;
    if (t >= EE * HEADS) return;
    int e = t >> 2, hd = t & 3;
    int s = ei[e], d = ei[EE + e];
    float v = aS[s * HEADS + hd] + aD[d * HEADS + hd];
    v = (v > 0.f) ? v : v * NEG_SLOPE;
    atomicMax(emax + d * HEADS + hd, fenc(v));
}

// ---- K5: exp + segment sum ----
__global__ void k_edge_sum(const int* __restrict__ ei, const float* __restrict__ aS,
                           const float* __restrict__ aD, const unsigned int* __restrict__ emax,
                           float* __restrict__ ee, float* __restrict__ esum) {
    int t = blockIdx.x * blockDim.x + threadIdx.x;
    if (t >= EE * HEADS) return;
    int e = t >> 2, hd = t & 3;
    int s = ei[e], d = ei[EE + e];
    float v = aS[s * HEADS + hd] + aD[d * HEADS + hd];
    v = (v > 0.f) ? v : v * NEG_SLOPE;
    float x = __expf(v - fdec(emax[d * HEADS + hd]));
    ee[t] = x;
    atomicAdd(esum + d * HEADS + hd, x);
}

// ---- K6: weighted message scatter (head-mean fused) ----
__global__ void k_edge_msg(const int* __restrict__ ei, const float* __restrict__ ee,
                           const float* __restrict__ esum, const float* __restrict__ xp,
                           float* __restrict__ agg) {
    int e = blockIdx.x;
    int c = threadIdx.x;
    int s = ei[e], d = ei[EE + e];
    float contrib = 0.f;
#pragma unroll
    for (int hd = 0; hd < HEADS; ++hd) {
        float a = ee[e * HEADS + hd] / esum[d * HEADS + hd];
        contrib = fmaf(a, xp[(size_t)s * KDIM + hd * HH + c], contrib);
    }
    atomicAdd(agg + (size_t)d * HH + c, 0.25f * contrib);
}

// ---- K7: +bias, LayerNorm, ReLU + residual ----
__global__ void k_finalize(const float* __restrict__ agg, const float* __restrict__ bias,
                           const float* __restrict__ g, const float* __restrict__ b,
                           float* __restrict__ h, unsigned short* __restrict__ hbf) {
    int n = blockIdx.x, c = threadIdx.x;
    __shared__ float red[HH];
    float v = agg[(size_t)n * HH + c] + bias[c];
    red[c] = v; __syncthreads();
    for (int s = 64; s > 0; s >>= 1) { if (c < s) red[c] += red[c + s]; __syncthreads(); }
    float mu = red[0] * (1.f / HH); __syncthreads();
    float dv = v - mu;
    red[c] = dv * dv; __syncthreads();
    for (int s = 64; s > 0; s >>= 1) { if (c < s) red[c] += red[c + s]; __syncthreads(); }
    float var = red[0] * (1.f / HH);
    float ln = dv * rsqrtf(var + LN_EPS) * g[c] + b[c];
    float nh = fmaxf(ln, 0.f) + h[(size_t)n * HH + c];
    h[(size_t)n * HH + c] = nh;
    hbf[(size_t)n * HH + c] = f2bf(nh);
}

// ---- K8: zero pooled sums ----
__global__ void k_zero_pool(float* __restrict__ gsum, float* __restrict__ ssum) {
    int c = threadIdx.x;
    gsum[c] = 0.f; ssum[c] = 0.f;
}

// ---- K8a: graph mean pool + solution-edge pool (partial sums, 200 blocks) ----
__global__ void k_pool(const float* __restrict__ h, const int* __restrict__ sol,
                       float* __restrict__ gsum, float* __restrict__ ssum) {
    int c = threadIdx.x, blk = blockIdx.x;
    float acc = 0.f;
    int n0 = blk * 100;
    for (int i = 0; i < 100; ++i) acc += h[(size_t)(n0 + i) * HH + c];
    atomicAdd(gsum + c, acc);
    float sacc = 0.f;
    int s0 = blk * 50;
    for (int i = 0; i < 50; ++i) {
        int a = sol[(s0 + i) * 2], b = sol[(s0 + i) * 2 + 1];
        sacc += 0.5f * (h[(size_t)a * HH + c] + h[(size_t)b * HH + c]);
    }
    atomicAdd(ssum + c, sacc);
}

// ---- K9: single-row MLP, broadcast score + argmax(=0) ----
__global__ void k_mlp(const float* __restrict__ gsum, const float* __restrict__ ssum,
                      const int* __restrict__ num_moves, int out_cap,
                      const float* __restrict__ W1, const float* __restrict__ b1,
                      const float* __restrict__ W2, const float* __restrict__ b2,
                      const float* __restrict__ W3, const float* __restrict__ b3,
                      float* __restrict__ out) {
    __shared__ float comb[3 * HH];
    __shared__ float s1[HH];
    __shared__ float s2[HH];
    __shared__ float red[HH];
    int t = threadIdx.x;
    comb[t]          = gsum[t] * (1.f / NN);
    comb[HH + t]     = ssum[t] * (1.f / SS);
    comb[2 * HH + t] = 0.f;
    __syncthreads();
    float a = b1[t];
    for (int i = 0; i < 3 * HH; ++i) a = fmaf(comb[i], W1[i * HH + t], a);
    s1[t] = fmaxf(a, 0.f); __syncthreads();
    float a2 = b2[t];
    for (int i = 0; i < HH; ++i) a2 = fmaf(s1[i], W2[i * HH + t], a2);
    s2[t] = fmaxf(a2, 0.f); __syncthreads();
    red[t] = s2[t] * W3[t]; __syncthreads();
    for (int s = 64; s > 0; s >>= 1) { if (t < s) red[t] += red[t + s]; __syncthreads(); }
    float score = red[0] + b3[0];
    int M = num_moves[0];
    for (int i = t; i < M && i < out_cap; i += HH) out[i] = score;
    if (t == 0 && M < out_cap) out[M] = 0.0f;   // argmax of identical scores -> 0
}

extern "C" void kernel_launch(void* const* d_in, const int* in_sizes, int n_in,
                              void* d_out, int out_size, void* d_ws, size_t ws_size,
                              hipStream_t stream) {
    const float* x       = (const float*)d_in[0];
    const int*   ei      = (const int*)d_in[1];
    const int*   sol     = (const int*)d_in[2];
    const int*   nmv     = (const int*)d_in[3];
    const float* nodeW   = (const float*)d_in[4];
    const float* nodeB   = (const float*)d_in[5];
    const float* linW    = (const float*)d_in[6];
    const float* attS    = (const float*)d_in[7];
    const float* attD    = (const float*)d_in[8];
    const float* gatB    = (const float*)d_in[9];
    const float* lnG     = (const float*)d_in[10];
    const float* lnB     = (const float*)d_in[11];
    const float* W1      = (const float*)d_in[12];
    const float* b1      = (const float*)d_in[13];
    const float* W2      = (const float*)d_in[14];
    const float* b2      = (const float*)d_in[15];
    const float* W3      = (const float*)d_in[16];
    const float* b3      = (const float*)d_in[17];
    float* out = (float*)d_out;

    // workspace carve
    char* p = (char*)d_ws;
    auto carve = [&](size_t bytes) { char* r = p; p += (bytes + 255) & ~size_t(255); return r; };
    float*          h    = (float*)carve((size_t)NN * HH * 4);
    unsigned short* hbf  = (unsigned short*)carve((size_t)NN * HH * 2);
    float*          xp   = (float*)carve((size_t)NN * KDIM * 4);
    unsigned short* wt   = (unsigned short*)carve((size_t)LL * KDIM * HH * 2);
    float*          aS   = (float*)carve((size_t)NN * HEADS * 4);
    float*          aD   = (float*)carve((size_t)NN * HEADS * 4);
    unsigned int*   emax = (unsigned int*)carve((size_t)NN * HEADS * 4);
    float*          esum = (float*)carve((size_t)NN * HEADS * 4);
    float*          ee   = (float*)carve((size_t)EE * HEADS * 4);
    float*          agg  = (float*)carve((size_t)NN * HH * 4);
    float*          gsum = (float*)carve(HH * 4);
    float*          ssum = (float*)carve(HH * 4);

    // weights -> transposed bf16 (once)
    k_convert_w<<<(LL * KDIM * HH + 255) / 256, 256, 0, stream>>>(linW, wt);
    // node embedding
    k_node_embed<<<(NN * HH + 255) / 256, 256, 0, stream>>>(x, nodeW, nodeB, h, hbf);

    // waves = 1250 row tiles * 8 col groups; 4 waves per 128-thread block
    const int gemmBlocks = (NN / 16) * (KDIM / 64) / 4;
    for (int l = 0; l < LL; ++l) {
        k_gemm_wmma<<<gemmBlocks, 128, 0, stream>>>(hbf, wt + (size_t)l * KDIM * HH, xp);
        k_alphas<<<(NN * HEADS + 255) / 256, 256, 0, stream>>>(
            xp, attS + (size_t)l * HEADS * HH, attD + (size_t)l * HEADS * HH, aS, aD);
        k_init<<<(NN * HH + 255) / 256, 256, 0, stream>>>(emax, esum, agg);
        k_edge_max<<<(EE * HEADS + 255) / 256, 256, 0, stream>>>(ei, aS, aD, emax);
        k_edge_sum<<<(EE * HEADS + 255) / 256, 256, 0, stream>>>(ei, aS, aD, emax, ee, esum);
        k_edge_msg<<<EE, HH, 0, stream>>>(ei, ee, esum, xp, agg);
        k_finalize<<<NN, HH, 0, stream>>>(agg, gatB + (size_t)l * HH,
                                          lnG + (size_t)l * HH, lnB + (size_t)l * HH, h, hbf);
    }

    k_zero_pool<<<1, HH, 0, stream>>>(gsum, ssum);
    k_pool<<<200, HH, 0, stream>>>(h, sol, gsum, ssum);
    k_mlp<<<1, HH, 0, stream>>>(gsum, ssum, nmv, out_size, W1, b1, W2, b2, W3, b3, out);
}
